// DualEncoderLSTMWeakSupervision_16415365005536
// MI455X (gfx1250) — matrix operand order repaired and checked by reading, not compiled, not run on hardware
//
#include <hip/hip_runtime.h>

// ---------------------------------------------------------------------------
// Dual-encoder LSTM with weak supervision, MI455X (gfx1250, wave32, WMMA).
//
// Pipeline (all on `stream`):
//   1. pack_w       : W fp32 [512,1024] -> bf16 WMMA B-fragment layout (1 MB)
//   2. lstm_kernel  : persistent per-16-rows LSTM, 50 steps, both encoders.
//                     GEMM via v_wmma_f32_16x16x32_bf16, gates in registers.
//                     512 threads = 16 waves, 4 tiles/wave (no VGPR spills).
//   3. qm_kernel    : qm = hq @ M        (v_wmma_f32_16x16x4_f32)
//   4. dist_kernel  : D  = qm @ hr^T     (v_wmma_f32_16x16x4_f32)
//   5. loss_partials/loss_final : deterministic hinge-loss reduction.
// ---------------------------------------------------------------------------

typedef __bf16 bf16_t;
typedef __attribute__((ext_vector_type(8)))  __bf16 v8bf;
typedef __attribute__((ext_vector_type(16))) __bf16 v16bf;
typedef __attribute__((ext_vector_type(8)))  float  v8f;
typedef __attribute__((ext_vector_type(2)))  float  v2f;

#define B_   512
#define T_   50
#define E_   256
#define H_   256
#define N4H  1024   // 4*H
// K dimension of [x_t, h] @ W is E_+H_ = 512 -> 16 chunks of 32 (bf16 WMMA)

__device__ __forceinline__ float sigm(float x) { return 1.0f / (1.0f + __expf(-x)); }

__device__ __forceinline__ v16bf cat8(v8bf lo, v8bf hi) {
  return __builtin_shufflevector(lo, hi, 0,1,2,3,4,5,6,7,8,9,10,11,12,13,14,15);
}

// ---------------------------------------------------------------------------
// 1. Pack W [k=0..511][n=0..1023] fp32 row-major into bf16 B-fragments.
// Assumed B layout for V_WMMA_F32_16X16X32_BF16 (32Kx16N, mirrors documented
// A layout): lane half h = lane>>4, element s holds K = 8h+s (s<8) or
// 16+8h+(s-8) (s>=8) of the 32-chunk, N = lane&15.
// One thread packs one (ntile, ktile, lane) -> 16 contiguous bf16 (32 B).
// ---------------------------------------------------------------------------
__global__ __launch_bounds__(256) void pack_w(const float* __restrict__ W,
                                              bf16_t* __restrict__ Wp) {
  int tid  = blockIdx.x * blockDim.x + threadIdx.x;  // 0..32767
  int lane = tid & 31;
  int kt   = (tid >> 5) & 15;   // k chunk 0..15
  int nt   = tid >> 9;          // n tile  0..63
  int n     = nt * 16 + (lane & 15);
  int kbase = kt * 32 + (lane >> 4) * 8;
  int out   = tid * 16;
#pragma unroll
  for (int j = 0; j < 8; ++j) {
    int k0 = kbase + 2 * j + (j >= 4 ? 8 : 0);
    Wp[out + 2*j]     = (bf16_t)W[(k0    ) * N4H + n];
    Wp[out + 2*j + 1] = (bf16_t)W[(k0 + 1) * N4H + n];
  }
}

// ---------------------------------------------------------------------------
// 2. Persistent LSTM encoder. grid = (32, 2): x = 16-row batch tile,
//    y = 0 queries / 1 replies. block = 512 threads = 16 waves.
// Wave w owns H-column block cb == w across all 4 gates (4 acc tiles), so the
// gate nonlinearity runs directly on the fp32 accumulator fragments and the
// per-wave register footprint stays spill-free (~80 VGPRs).
// ---------------------------------------------------------------------------
__global__ __launch_bounds__(512) void lstm_kernel(
    const int*   __restrict__ tok_q, const int*   __restrict__ tok_r,
    const int*   __restrict__ len_q, const int*   __restrict__ len_r,
    const float* __restrict__ emb,   const bf16_t* __restrict__ Wp,
    const float* __restrict__ bias,
    const float* __restrict__ p_i, const float* __restrict__ p_f,
    const float* __restrict__ p_o,
    float* __restrict__ hq, float* __restrict__ hr)
{
  __shared__ __align__(16) bf16_t aT[16][520];  // [x_t | h] bf16, padded stride
  __shared__ float cT[16][260];                 // cell state fp32, padded
  __shared__ float hT[16][256];                 // hidden fp32 (for final output)
  __shared__ float piS[256], pfS[256], poS[256], bS[N4H];
  __shared__ int lenS[16];

  const int tid  = threadIdx.x;
  const int wv   = tid >> 5;      // wave 0..15  == H column block cb
  const int lane = tid & 31;
  const int lh   = lane >> 4;
  const int ln   = lane & 15;

  const bool isQ = (blockIdx.y == 0);
  const int*   tok  = isQ ? tok_q : tok_r;
  const int*   len  = isQ ? len_q : len_r;
  float*       hout = isQ ? hq : hr;

  // ---- init shared state (512 threads: 8 elements each) -------------------
  {
    int r  = tid >> 5;            // 0..15
    int c0 = (tid & 31) << 3;     // 0..248
    if (tid < 256) {
      piS[tid] = p_i[tid];
      pfS[tid] = p_f[tid];
      poS[tid] = p_o[tid];
    }
    bS[tid]       = bias[tid];
    bS[tid + 512] = bias[tid + 512];
    if (tid < 16) lenS[tid] = len[blockIdx.x * 16 + tid];
#pragma unroll
    for (int u = 0; u < 8; ++u) {
      aT[r][256 + c0 + u] = (bf16_t)0.0f;   // h part of A tile
      cT[r][c0 + u] = 0.0f;
      hT[r][c0 + u] = 0.0f;
    }
  }

  for (int t = 0; t < T_; ++t) {
    __syncthreads();
    // ---- gather x_t = embeddings[tokens[:, t]] -> bf16 A tile -------------
    {
      int r  = tid >> 5;
      int c0 = (tid & 31) << 3;
      int token = tok[(blockIdx.x * 16 + r) * T_ + t];
      const float* er = emb + (long long)token * E_ + c0;
#pragma unroll
      for (int u = 0; u < 8; ++u) aT[r][c0 + u] = (bf16_t)er[u];
    }
    __syncthreads();

    // ---- z = [x_t, h] @ W + b : 4 accumulator tiles (one per gate) --------
    v8f acc[4];
#pragma unroll
    for (int g = 0; g < 4; ++g) {
      float bb = bS[g * 256 + wv * 16 + ln];
      acc[g] = (v8f){bb, bb, bb, bb, bb, bb, bb, bb};
    }
#pragma unroll 2
    for (int kk = 0; kk < 16; ++kk) {
      // A fragment (16x32 bf16): lane half selects K sub-range (ISA layout)
      v8bf a0 = *(const v8bf*)&aT[ln][kk * 32 + lh * 8];
      v8bf a1 = *(const v8bf*)&aT[ln][kk * 32 + lh * 8 + 16];
      v16bf afrag = cat8(a0, a1);
#pragma unroll
      for (int g = 0; g < 4; ++g) {
        int ntile = g * 16 + wv;
        const bf16_t* bp = Wp + (((ntile * 16 + kk) * 32 + lane) << 4);
        v8bf b0 = *(const v8bf*)bp;
        v8bf b1 = *(const v8bf*)(bp + 8);
        v16bf bfrag = cat8(b0, b1);
        acc[g] = __builtin_amdgcn_wmma_f32_16x16x32_bf16(
            false, afrag, false, bfrag, (short)0, acc[g], false, false);
      }
    }
    __syncthreads();  // all waves done reading aT (h part) before rewriting it

    // ---- gates in registers; C/D layout: row = v + 8*lh, col = wv*16+ln ---
    {
      int col = wv * 16 + ln;
      float pi = piS[col], pf = pfS[col], po = poS[col];
#pragma unroll
      for (int v = 0; v < 8; ++v) {
        int row = v + 8 * lh;
        float cc = cT[row][col];
        float iv = sigm(acc[0][v] + pi * cc);
        float jv = tanhf(acc[1][v]);
        float fv = sigm(acc[2][v] + 2.0f + pf * cc);   // forget_bias = 2.0
        float cn = fv * cc + iv * jv;
        float ov = sigm(acc[3][v] + po * cn);
        float hn = ov * tanhf(cn);
        if (t < lenS[row]) {                 // dynamic_rnn: freeze past length
          cT[row][col] = cn;
          hT[row][col] = hn;
          aT[row][256 + col] = (bf16_t)hn;
        }
      }
    }
  }

  __syncthreads();
  {
    int r  = tid >> 5;
    int c0 = (tid & 31) << 3;
    float* out = hout + (blockIdx.x * 16 + r) * H_ + c0;
#pragma unroll
    for (int u = 0; u < 8; ++u) out[u] = hT[r][c0 + u];
  }
}

// ---------------------------------------------------------------------------
// fp32 WMMA 16x16x4 helpers (head GEMMs, exact fp32).
// Assumed layouts: A 16x4: M=lane&15, VGPR0/1 = K = 2*(lane>>4)+{0,1}.
//                  B 4x16: N=lane&15, VGPR0/1 = K = 2*(lane>>4)+{0,1}.
// ---------------------------------------------------------------------------
__global__ __launch_bounds__(256) void qm_kernel(const float* __restrict__ hqv,
                                                 const float* __restrict__ Mw,
                                                 float* __restrict__ qm) {
  int wv = threadIdx.x >> 5, lane = threadIdx.x & 31, lh = lane >> 4, ln = lane & 15;
  int tile = blockIdx.x * 8 + wv;   // 0..511
  int mt = tile >> 4, nt = tile & 15;
  v8f acc = (v8f){0, 0, 0, 0, 0, 0, 0, 0};
  const float* arow = hqv + (mt * 16 + ln) * H_;
  for (int kb = 0; kb < 64; ++kb) {
    int kc = kb * 4 + 2 * lh;
    v2f a = {arow[kc], arow[kc + 1]};
    v2f b = {Mw[kc * H_ + nt * 16 + ln], Mw[(kc + 1) * H_ + nt * 16 + ln]};
    acc = __builtin_amdgcn_wmma_f32_16x16x4_f32(false, a, false, b, (short)0,
                                                acc, false, false);
  }
#pragma unroll
  for (int v = 0; v < 8; ++v)
    qm[(mt * 16 + v + 8 * lh) * H_ + nt * 16 + ln] = acc[v];
}

__global__ __launch_bounds__(256) void dist_kernel(const float* __restrict__ qm,
                                                   const float* __restrict__ hrv,
                                                   float* __restrict__ D) {
  int wv = threadIdx.x >> 5, lane = threadIdx.x & 31, lh = lane >> 4, ln = lane & 15;
  int tile = blockIdx.x * 8 + wv;   // 0..1023
  int mt = tile >> 5, nt = tile & 31;
  v8f acc = (v8f){0, 0, 0, 0, 0, 0, 0, 0};
  const float* arow = qm  + (mt * 16 + ln) * H_;
  const float* brow = hrv + (nt * 16 + ln) * H_;   // B = hr^T -> row of hr
  for (int kb = 0; kb < 64; ++kb) {
    int kc = kb * 4 + 2 * lh;
    v2f a = {arow[kc], arow[kc + 1]};
    v2f b = {brow[kc], brow[kc + 1]};
    acc = __builtin_amdgcn_wmma_f32_16x16x4_f32(false, a, false, b, (short)0,
                                                acc, false, false);
  }
#pragma unroll
  for (int v = 0; v < 8; ++v)
    D[(mt * 16 + v + 8 * lh) * B_ + nt * 16 + ln] = acc[v];
}

// ---------------------------------------------------------------------------
// Deterministic hinge-loss reduction (no float atomics).
// ---------------------------------------------------------------------------
__global__ __launch_bounds__(256) void loss_partials(const float* __restrict__ D,
                                                     const float* __restrict__ wd,
                                                     float* __restrict__ partials) {
  __shared__ float red[256];
  float s = 0.0f;
  for (int idx = blockIdx.x * 256 + threadIdx.x; idx < B_ * B_; idx += 256 * 256) {
    int j = idx & (B_ - 1);
    float d   = D[idx];
    float pos = D[j * (B_ + 1)];
    float w   = wd[idx];
    float wp  = wd[j * (B_ + 1)];
    float wn  = fmaxf(0.0f, w / wp - 1.0f);
    s += fmaxf(0.0f, d - pos + wn);
  }
  red[threadIdx.x] = s;
  __syncthreads();
  for (int off = 128; off > 0; off >>= 1) {
    if (threadIdx.x < off) red[threadIdx.x] += red[threadIdx.x + off];
    __syncthreads();
  }
  if (threadIdx.x == 0) partials[blockIdx.x] = red[0];
}

__global__ __launch_bounds__(256) void loss_final(const float* __restrict__ partials,
                                                  float* __restrict__ out) {
  __shared__ float red[256];
  red[threadIdx.x] = partials[threadIdx.x];
  __syncthreads();
  for (int off = 128; off > 0; off >>= 1) {
    if (threadIdx.x < off) red[threadIdx.x] += red[threadIdx.x + off];
    __syncthreads();
  }
  if (threadIdx.x == 0) out[0] = red[0];
}

// ---------------------------------------------------------------------------
extern "C" void kernel_launch(void* const* d_in, const int* in_sizes, int n_in,
                              void* d_out, int out_size, void* d_ws, size_t ws_size,
                              hipStream_t stream) {
  (void)in_sizes; (void)n_in; (void)out_size; (void)ws_size;
  const int*   tok_q = (const int*)d_in[0];
  const int*   tok_r = (const int*)d_in[1];
  const int*   len_q = (const int*)d_in[2];
  const int*   len_r = (const int*)d_in[3];
  const float* wd    = (const float*)d_in[4];
  const float* emb   = (const float*)d_in[5];
  const float* W     = (const float*)d_in[6];
  const float* bias  = (const float*)d_in[7];
  const float* p_i   = (const float*)d_in[8];
  const float* p_f   = (const float*)d_in[9];
  const float* p_o   = (const float*)d_in[10];
  const float* Mw    = (const float*)d_in[11];
  float* out = (float*)d_out;

  char* w = (char*)d_ws;
  bf16_t* Wp = (bf16_t*)w;                           // 512*1024*2 = 1 MB
  float* hq  = (float*)(w + (1u << 20));             // 512*256 fp32
  float* hr  = hq + B_ * H_;
  float* qm  = hr + B_ * H_;
  float* D   = qm + B_ * H_;                         // 512*512 fp32
  float* partials = D + B_ * B_;                     // 256 fp32

  pack_w<<<128, 256, 0, stream>>>(W, Wp);
  lstm_kernel<<<dim3(32, 2), 512, 0, stream>>>(tok_q, tok_r, len_q, len_r, emb,
                                               Wp, bias, p_i, p_f, p_o, hq, hr);
  qm_kernel<<<64, 256, 0, stream>>>(hq, Mw, qm);
  dist_kernel<<<128, 256, 0, stream>>>(qm, hr, D);
  loss_partials<<<256, 256, 0, stream>>>(D, wd, partials);
  loss_final<<<1, 256, 0, stream>>>(partials, out);
}
